// SpectralAttention_137438953633
// MI455X (gfx1250) — matrix operand order repaired and checked by reading, not compiled, not run on hardware
//
#include <hip/hip_runtime.h>
#include <hip/hip_bf16.h>
#include <math.h>
#include <stdint.h>

// Problem sizes (fixed by the reference)
#define BB 2
#define TT 2048
#define CC 1024
#define HH 16
#define HD 64
#define MM (BB * TT)   // 4096 rows

typedef __attribute__((ext_vector_type(16))) __bf16          v16bf;
typedef __attribute__((ext_vector_type(8)))  float           v8f;
typedef __attribute__((ext_vector_type(8)))  unsigned int    v8u;
typedef __attribute__((ext_vector_type(8)))  unsigned short  u16x8;
typedef __attribute__((ext_vector_type(4)))  unsigned int    u32x4;
typedef __attribute__((ext_vector_type(8)))  int             i32x8;
typedef __attribute__((ext_vector_type(4)))  int             i32x4;

#if defined(__has_builtin)
#if __has_builtin(__builtin_amdgcn_tensor_load_to_lds) && \
    __has_builtin(__builtin_amdgcn_s_wait_tensorcnt)
#define USE_TDM 1
#endif
#endif
#ifndef USE_TDM
#define USE_TDM 0
#endif

// ---------------------------------------------------------------- helpers ---

__device__ __forceinline__ unsigned short f2bf(float f) {
  union { float f; unsigned int u; } x; x.f = f;
  unsigned int u = x.u;
  unsigned int r = u + 0x7FFFu + ((u >> 16) & 1u);   // round-to-nearest-even
  return (unsigned short)(r >> 16);
}

__device__ __forceinline__ v8f zero8() {
  v8f z;
#pragma unroll
  for (int i = 0; i < 8; i++) z[i] = 0.0f;
  return z;
}

// Generic (flat) LDS pointer -> LDS byte offset (aperture keeps offset in [31:0])
__device__ __forceinline__ unsigned lds_off(const void* p) {
  return (unsigned)(uintptr_t)p;
}

#if USE_TDM
// Tensor Data Mover: 2D tile load Global -> LDS (async, TENSORcnt-tracked).
// D# packing per CDNA5 ISA.
// 6-arg builtin: (uint32x4 g0, int32x8 g1, int32x4 g2, int32x4 g3, int32x8, i32)
__device__ __forceinline__ void tdm_load_2d(unsigned lds_byte_off, const void* g,
                                            unsigned tile_d0, unsigned tile_d1,
                                            unsigned long long stride_elems) {
  unsigned long long ga = (unsigned long long)g;
  u32x4 g0;
  g0[0] = 1u;                                            // count=1, no gather
  g0[1] = lds_byte_off;                                  // lds_addr
  g0[2] = (unsigned)ga;                                  // global_addr[31:0]
  g0[3] = (unsigned)((ga >> 32) & 0x01FFFFFFu)           // global_addr[56:32]
        | (2u << 30);                                    // type = 2 ("image")
  unsigned long long td0 = stride_elems;                 // tensor_dim0 >= tile w
  unsigned td1 = 0x40000000u;                            // rows: effectively inf
  i32x8 g1;
  g1[0] = (int)(1u << 16);                               // data_size=1 -> 2 bytes
  g1[1] = (int)((unsigned)(td0 & 0xFFFFull) << 16);      // tensor_dim0[15:0]
  g1[2] = (int)((unsigned)((td0 >> 16) & 0xFFFFull)      // tensor_dim0[31:16]
        | ((td1 & 0xFFFFu) << 16));                      // tensor_dim1[15:0]
  g1[3] = (int)(((td1 >> 16) & 0xFFFFu)                  // tensor_dim1[31:16]
        | ((tile_d0 & 0xFFFFu) << 16));                  // tile_dim0
  g1[4] = (int)(tile_d1 & 0xFFFFu);                      // tile_dim1 (tile_dim2=0)
  g1[5] = (int)(unsigned)(stride_elems & 0xFFFFFFFFull); // dim0_stride[31:0]
  g1[6] = (int)(unsigned)((stride_elems >> 32) & 0xFFFFull);
  g1[7] = 0;
  i32x4 z4; z4[0] = 0; z4[1] = 0; z4[2] = 0; z4[3] = 0;
  i32x8 z8;
#pragma unroll
  for (int i = 0; i < 8; i++) z8[i] = 0;
  __builtin_amdgcn_tensor_load_to_lds(g0, g1, z4, z4, z8, 0);
}
#endif

// A-fragment (16x32 bf16, row-major source with row stride `ld` elements).
__device__ __forceinline__ v16bf load_a_frag(const unsigned short* p, int ld) {
  int lane = threadIdx.x & 31;
  int half = lane >> 4, m = lane & 15;
  const unsigned short* rp = p + m * ld;
  v8u r;
#pragma unroll
  for (int i = 0; i < 8; i++) {
    int kb = (i < 4) ? (half * 8 + 2 * i) : (16 + half * 8 + 2 * (i - 4));
    r[i] = *(const unsigned int*)(rp + kb);
  }
  return __builtin_bit_cast(v16bf, r);
}

// B-fragment (32x16 bf16) from an [n][k] tile (K pairs contiguous), stride `ld`.
__device__ __forceinline__ v16bf load_b_frag(const unsigned short* p, int ld) {
  int lane = threadIdx.x & 31;
  int half = lane >> 4, n = lane & 15;
  const unsigned short* cp = p + n * ld;
  v8u r;
#pragma unroll
  for (int i = 0; i < 8; i++) {
    int kb = (i < 4) ? (half * 8 + 2 * i) : (16 + half * 8 + 2 * (i - 4));
    r[i] = *(const unsigned int*)(cp + kb);
  }
  return __builtin_bit_cast(v16bf, r);
}

__device__ __forceinline__ v8f wmma_bf16(v16bf a, v16bf b, v8f c) {
  return __builtin_amdgcn_wmma_f32_16x16x32_bf16(
      false, a, false, b, (short)0, c, false, false);
}

// ---------------------------------------------------------------- kernels ---

__global__ __launch_bounds__(256) void energy_kernel(
    const float* __restrict__ x, float* __restrict__ e) {
  __shared__ float red[8];
  int row = blockIdx.x, tid = threadIdx.x;
  const float* p = x + (size_t)row * CC;
  float s = 0.0f;
  for (int c = tid; c < CC; c += 256) { float v = p[c]; s += v * v; }
#pragma unroll
  for (int m = 16; m >= 1; m >>= 1) s += __shfl_xor(s, m);
  if ((tid & 31) == 0) red[tid >> 5] = s;
  __syncthreads();
  if (tid == 0) {
    float t = 0.0f;
#pragma unroll
    for (int i = 0; i < 8; i++) t += red[i];
    e[row] = sqrtf(t);
  }
}

__global__ __launch_bounds__(256) void f32_to_bf16_kernel(
    const float* __restrict__ in, unsigned short* __restrict__ out, int n) {
  int i = blockIdx.x * 256 + threadIdx.x;
  if (i < n) out[i] = f2bf(in[i]);
}

// Y[M x 1024] = A[M x 1024] @ W[1024 x 1024] + bias   (bf16 in, f32 out)
// 256 threads = 8 waves; block tile 128x128; wave tile 32x64 (2x4 WMMA tiles).
// Double-buffered LDS; A tile via async TDM, W tile pipelined through registers.
__global__ __launch_bounds__(256) void gemm_bf16_kernel(
    const unsigned short* __restrict__ A, const unsigned short* __restrict__ W,
    const float* __restrict__ bias, float* __restrict__ Y) {
  __shared__ __align__(16) unsigned short sX [2][128 * 32];  // [m][k]
  __shared__ __align__(16) unsigned short sWt[2][128 * 32];  // [n][k] transposed
  int tid = threadIdx.x, w = tid >> 5;
  int lane = tid & 31, half = lane >> 4, n0 = lane & 15;
  int msub = (w >> 1) * 32, nsub = (w & 1) * 64;
  int mblk = blockIdx.x * 128, nblk = blockIdx.y * 128;

  // per-thread W staging coordinates (two 16B chunks per tile)
  int cW0 = tid >> 4,          nW = (tid & 15) * 8;
  int cW1 = cW0 + 16;
  // per-thread A staging coordinates (fallback path)
  int rA0 = tid >> 2,          cA0 = (tid & 3) * 8;
  int rA1 = (tid + 256) >> 2,  cA1 = ((tid + 256) & 3) * 8;

  v8f acc[2][4];
#pragma unroll
  for (int ai = 0; ai < 2; ai++)
#pragma unroll
    for (int bi = 0; bi < 4; bi++) acc[ai][bi] = zero8();

  // ---- prologue: stage tile 0 into buffer 0 (synchronous)
#if USE_TDM
  if (tid < 32)
    tdm_load_2d(lds_off(sX[0]), A + (size_t)mblk * CC, 32, 128, CC);
#else
  *(u16x8*)&sX[0][rA0 * 32 + cA0] =
      *(const u16x8*)&A[(size_t)(mblk + rA0) * CC + cA0];
  *(u16x8*)&sX[0][rA1 * 32 + cA1] =
      *(const u16x8*)&A[(size_t)(mblk + rA1) * CC + cA1];
#endif
  {
    u16x8 p0 = *(const u16x8*)&W[(size_t)cW0 * CC + nblk + nW];
    u16x8 p1 = *(const u16x8*)&W[(size_t)cW1 * CC + nblk + nW];
#pragma unroll
    for (int i = 0; i < 8; i++) {
      sWt[0][(nW + i) * 32 + cW0] = p0[i];
      sWt[0][(nW + i) * 32 + cW1] = p1[i];
    }
  }

  for (int ks = 0; ks < 32; ks++) {
    int cur = ks & 1, nxt = cur ^ 1;
    int kbn = (ks + 1) * 32;
    bool pre = (ks + 1 < 32);
    __syncthreads();   // (a) everyone done computing on buffer `nxt`

    u16x8 w0 = {}, w1 = {};
#if !USE_TDM
    u16x8 a0 = {}, a1 = {};
#endif
    if (pre) {
#if USE_TDM
      if (tid < 32)   // async prefetch of next A tile
        tdm_load_2d(lds_off(sX[nxt]), A + (size_t)mblk * CC + kbn, 32, 128, CC);
#else
      a0 = *(const u16x8*)&A[(size_t)(mblk + rA0) * CC + kbn + cA0];
      a1 = *(const u16x8*)&A[(size_t)(mblk + rA1) * CC + kbn + cA1];
#endif
      w0 = *(const u16x8*)&W[(size_t)(kbn + cW0) * CC + nblk + nW];
      w1 = *(const u16x8*)&W[(size_t)(kbn + cW1) * CC + nblk + nW];
    }
#if USE_TDM
    if (tid < 32) {   // tile ks complete; tile ks+1 may stay in flight
      if (pre) __builtin_amdgcn_s_wait_tensorcnt(1);
      else     __builtin_amdgcn_s_wait_tensorcnt(0);
    }
#endif
    __syncthreads();   // (b) buffer `cur` fully staged and visible

    v16bf f0 = load_a_frag(sX[cur] + msub * 32, 32);
    v16bf f1 = load_a_frag(sX[cur] + (msub + 16) * 32, 32);
#pragma unroll
    for (int bi = 0; bi < 4; bi++) {
      v16bf b = load_b_frag(sWt[cur] + (nsub + bi * 16) * 32, 32);
      acc[0][bi] = wmma_bf16(f0, b, acc[0][bi]);
      acc[1][bi] = wmma_bf16(f1, b, acc[1][bi]);
    }

    if (pre) {   // drain register-staged chunks after compute (latency hidden)
#pragma unroll
      for (int i = 0; i < 8; i++) {
        sWt[nxt][(nW + i) * 32 + cW0] = w0[i];
        sWt[nxt][(nW + i) * 32 + cW1] = w1[i];
      }
#if !USE_TDM
      *(u16x8*)&sX[nxt][rA0 * 32 + cA0] = a0;
      *(u16x8*)&sX[nxt][rA1 * 32 + cA1] = a1;
#endif
    }
  }

#pragma unroll
  for (int ai = 0; ai < 2; ai++)
#pragma unroll
    for (int bi = 0; bi < 4; bi++)
#pragma unroll
      for (int g = 0; g < 8; g++) {
        int row = mblk + msub + ai * 16 + g + 8 * half;
        int col = nblk + nsub + bi * 16 + n0;
        Y[(size_t)row * CC + col] = acc[ai][bi][g] + bias[col];
      }
}

// 2048-pt radix-2 DIT FFT of one (b,h,d) series + spectral filter.
// Writes bf16 real (and optionally imag) in [B,H,T,HD] layout.
__global__ __launch_bounds__(128) void fft_filter_kernel(
    const float* __restrict__ X, unsigned short* __restrict__ ore,
    unsigned short* __restrict__ oim, const float* __restrict__ alpha,
    const float* __restrict__ fas, const float* __restrict__ fd) {
  __shared__ float sre[TT];
  __shared__ float sim[TT];
  int bi = blockIdx.x;
  int d = bi & 63, h = (bi >> 6) & 15, b = bi >> 10;
  int tid = threadIdx.x;

  const float* src = X + (size_t)b * TT * CC + h * HD + d;
  for (int i = tid; i < TT; i += 128) {
    int j = (int)(__brev((unsigned)i) >> 21);  // 11-bit reversal
    sre[i] = src[(size_t)j * CC];
    sim[i] = 0.0f;
  }
  __syncthreads();

  for (int len = 2; len <= TT; len <<= 1) {
    int hl = len >> 1;
    float base = -6.283185307179586f / (float)len;
    for (int k = tid; k < (TT / 2); k += 128) {
      int pos = k & (hl - 1);
      int i0 = ((k & ~(hl - 1)) << 1) + pos;
      int i1 = i0 + hl;
      float ang = base * (float)pos;
      float wr = __cosf(ang), wi = __sinf(ang);
      float xr = sre[i1], xi = sim[i1];
      float tr = wr * xr - wi * xi, ti = wr * xi + wi * xr;
      float ur = sre[i0], ui = sim[i0];
      sre[i1] = ur - tr; sim[i1] = ui - ti;
      sre[i0] = ur + tr; sim[i0] = ui + ti;
    }
    __syncthreads();
  }

  float aa = alpha[0] + fas[0] * (fd[0] - 1.5f);
  size_t obase = ((size_t)(b * HH + h)) * TT * HD + d;
  for (int t = tid; t < TT; t += 128) {
    float fr = (t < TT / 2 ? (float)t : (float)(t - TT)) * (1.0f / (float)TT);
    float ph = aa * atanf(logf(fabsf(fr) + 1e-10f));
    float cp = __cosf(ph), sp = __sinf(ph);
    float re = sre[t], im = sim[t];
    ore[obase + (size_t)t * HD] = f2bf(re * cp - im * sp);
    if (oim) oim[obase + (size_t)t * HD] = f2bf(re * sp + im * cp);
  }
}

// Flash attention over spectral Q/K/V. grid = (B*H, T/128); 8 waves/block.
// Double-buffered K/V tiles: K via async TDM, V pipelined through registers.
__global__ __launch_bounds__(256) void flash_attn_kernel(
    const unsigned short* __restrict__ qr, const unsigned short* __restrict__ qi,
    const unsigned short* __restrict__ kr, const unsigned short* __restrict__ ki,
    const unsigned short* __restrict__ vr, float* __restrict__ attended) {
  __shared__ __align__(16) unsigned short sKr[2][32 * HD];  // [key][d]
  __shared__ __align__(16) unsigned short sKi[2][32 * HD];  // [key][d]
  __shared__ __align__(16) unsigned short sVt[2][HD * 32];  // [d][key]
  __shared__ __align__(16) unsigned short sP [8][16 * 32];

  int tid = threadIdx.x, w = tid >> 5;
  int lane = tid & 31, half = lane >> 4, n0 = lane & 15;
  int bh = blockIdx.x, b = bh >> 4, h = bh & 15;
  int t0 = blockIdx.y * 128 + w * 16;
  size_t headoff = (size_t)bh * TT * HD;
  int key = tid >> 3, d8 = (tid & 7) * 8;   // staging coordinates

  const unsigned short* qrb = qr + headoff + (size_t)t0 * HD;
  const unsigned short* qib = qi + headoff + (size_t)t0 * HD;
  v16bf aQr0 = load_a_frag(qrb, HD), aQr1 = load_a_frag(qrb + 32, HD);
  v16bf aQi0 = load_a_frag(qib, HD), aQi1 = load_a_frag(qib + 32, HD);

  float mrow[8], lrow[8];
#pragma unroll
  for (int g = 0; g < 8; g++) { mrow[g] = -1e30f; lrow[g] = 0.0f; }
  v8f O[4];
#pragma unroll
  for (int c = 0; c < 4; c++) O[c] = zero8();

  // ---- prologue: stage key-tile 0 into buffer 0
#if USE_TDM
  if (tid < 32) {
    tdm_load_2d(lds_off(sKr[0]), kr + headoff, HD, 32, HD);
    tdm_load_2d(lds_off(sKi[0]), ki + headoff, HD, 32, HD);
  }
#else
  *(u16x8*)&sKr[0][key * HD + d8] =
      *(const u16x8*)&kr[headoff + (size_t)key * HD + d8];
  *(u16x8*)&sKi[0][key * HD + d8] =
      *(const u16x8*)&ki[headoff + (size_t)key * HD + d8];
#endif
  {
    u16x8 v = *(const u16x8*)&vr[headoff + (size_t)key * HD + d8];
#pragma unroll
    for (int i = 0; i < 8; i++) sVt[0][(d8 + i) * 32 + key] = v[i];
  }

  for (int j = 0; j < TT / 32; j++) {
    int cur = j & 1, nxt = cur ^ 1;
    int kbn = (j + 1) * 32;
    bool pre = (j + 1 < TT / 32);
    __syncthreads();   // (a) buffer `nxt` free for restaging

    u16x8 vreg = {};
#if !USE_TDM
    u16x8 krr = {}, kir = {};
#endif
    if (pre) {
#if USE_TDM
      if (tid < 32) {
        tdm_load_2d(lds_off(sKr[nxt]), kr + headoff + (size_t)kbn * HD, HD, 32, HD);
        tdm_load_2d(lds_off(sKi[nxt]), ki + headoff + (size_t)kbn * HD, HD, 32, HD);
      }
#else
      krr = *(const u16x8*)&kr[headoff + (size_t)(kbn + key) * HD + d8];
      kir = *(const u16x8*)&ki[headoff + (size_t)(kbn + key) * HD + d8];
#endif
      vreg = *(const u16x8*)&vr[headoff + (size_t)(kbn + key) * HD + d8];
    }
#if USE_TDM
    if (tid < 32) {   // current tile's two TDMs complete; next pair in flight
      if (pre) __builtin_amdgcn_s_wait_tensorcnt(2);
      else     __builtin_amdgcn_s_wait_tensorcnt(0);
    }
#endif
    __syncthreads();   // (b) buffer `cur` fully staged and visible

    // S = (Qr Kr^T - Qi Ki^T) / sqrt(HD)
    v8f S[2];
#pragma unroll
    for (int s = 0; s < 2; s++) {
      v8f aR = zero8(), aI = zero8();
      aR = wmma_bf16(aQr0, load_b_frag(sKr[cur] + s * 16 * HD,      HD), aR);
      aR = wmma_bf16(aQr1, load_b_frag(sKr[cur] + s * 16 * HD + 32, HD), aR);
      aI = wmma_bf16(aQi0, load_b_frag(sKi[cur] + s * 16 * HD,      HD), aI);
      aI = wmma_bf16(aQi1, load_b_frag(sKi[cur] + s * 16 * HD + 32, HD), aI);
      S[s] = (aR - aI) * 0.125f;
    }

    // online softmax (row reductions stay within each 16-lane half)
#pragma unroll
    for (int g = 0; g < 8; g++) {
      float mx = fmaxf(S[0][g], S[1][g]);
      mx = fmaxf(mx, __shfl_xor(mx, 1));
      mx = fmaxf(mx, __shfl_xor(mx, 2));
      mx = fmaxf(mx, __shfl_xor(mx, 4));
      mx = fmaxf(mx, __shfl_xor(mx, 8));
      float mnew  = fmaxf(mrow[g], mx);
      float scale = __expf(mrow[g] - mnew);
      float p0 = __expf(S[0][g] - mnew);
      float p1 = __expf(S[1][g] - mnew);
      float rs = p0 + p1;
      rs += __shfl_xor(rs, 1);
      rs += __shfl_xor(rs, 2);
      rs += __shfl_xor(rs, 4);
      rs += __shfl_xor(rs, 8);
      lrow[g] = lrow[g] * scale + rs;
      mrow[g] = mnew;
      O[0][g] *= scale; O[1][g] *= scale; O[2][g] *= scale; O[3][g] *= scale;
      int row = g + 8 * half;
      sP[w][row * 32 + n0]      = f2bf(p0);
      sP[w][row * 32 + 16 + n0] = f2bf(p1);
    }

    v16bf aP = load_a_frag(sP[w], 32);
#pragma unroll
    for (int c = 0; c < 4; c++)
      O[c] = wmma_bf16(aP, load_b_frag(sVt[cur] + c * 16 * 32, 32), O[c]);

    if (pre) {   // drain register-staged tiles after compute
#pragma unroll
      for (int i = 0; i < 8; i++) sVt[nxt][(d8 + i) * 32 + key] = vreg[i];
#if !USE_TDM
      *(u16x8*)&sKr[nxt][key * HD + d8] = krr;
      *(u16x8*)&sKi[nxt][key * HD + d8] = kir;
#endif
    }
  }

#pragma unroll
  for (int g = 0; g < 8; g++) {
    float inv = 1.0f / lrow[g];
    int trow = t0 + g + 8 * half;
    float* dst = attended + ((size_t)(b * TT + trow)) * CC + h * HD;
#pragma unroll
    for (int c = 0; c < 4; c++) dst[c * 16 + n0] = O[c][g] * inv;
  }
}

__global__ __launch_bounds__(256) void final_scale_kernel(
    const float* __restrict__ ypre, const float* __restrict__ energy,
    const float* __restrict__ en, float* __restrict__ out) {
  __shared__ float red[8];
  __shared__ float sc;
  int row = blockIdx.x, tid = threadIdx.x;
  const float* p = ypre + (size_t)row * CC;
  float s = 0.0f;
  for (int c = tid; c < CC; c += 256) { float v = p[c]; s += v * v; }
#pragma unroll
  for (int m = 16; m >= 1; m >>= 1) s += __shfl_xor(s, m);
  if ((tid & 31) == 0) red[tid >> 5] = s;
  __syncthreads();
  if (tid == 0) {
    float t = 0.0f;
#pragma unroll
    for (int i = 0; i < 8; i++) t += red[i];
    sc = energy[row] / (sqrtf(t) + 1e-8f) * en[0];
  }
  __syncthreads();
  float k = sc;
  for (int c = tid; c < CC; c += 256)
    out[(size_t)row * CC + c] = p[c] * k;
}

// ----------------------------------------------------------------- launch ---

extern "C" void kernel_launch(void* const* d_in, const int* in_sizes, int n_in,
                              void* d_out, int out_size, void* d_ws, size_t ws_size,
                              hipStream_t stream) {
  (void)in_sizes; (void)n_in; (void)out_size; (void)ws_size;

  const float* x    = (const float*)d_in[0];
  const float* fd   = (const float*)d_in[1];
  const float* wq   = (const float*)d_in[2];
  const float* bq   = (const float*)d_in[3];
  const float* wk   = (const float*)d_in[4];
  const float* bk   = (const float*)d_in[5];
  const float* wv   = (const float*)d_in[6];
  const float* bv   = (const float*)d_in[7];
  const float* wo   = (const float*)d_in[8];
  const float* bo   = (const float*)d_in[9];
  const float* alpha = (const float*)d_in[10];
  const float* fas   = (const float*)d_in[11];
  const float* enrm  = (const float*)d_in[12];
  float* out = (float*)d_out;

  char* ws = (char*)d_ws;
  const size_t MB = 1ull << 20;
  unsigned short* xbf   = (unsigned short*)(ws);             // 8 MB (reused)
  unsigned short* wbf0  = (unsigned short*)(ws + 8 * MB);
  unsigned short* wbf1  = (unsigned short*)(ws + 10 * MB);
  unsigned short* wbf2  = (unsigned short*)(ws + 12 * MB);
  unsigned short* wbf3  = (unsigned short*)(ws + 14 * MB);
  float*          qf32  = (float*)(ws + 16 * MB);            // 16 MB (reused)
  float*          kf32  = (float*)(ws + 32 * MB);            // 16 MB (reused)
  float*          vf32  = (float*)(ws + 48 * MB);            // 16 MB
  unsigned short* qfr   = (unsigned short*)(ws + 64 * MB);   // 8 MB each
  unsigned short* qfi   = (unsigned short*)(ws + 72 * MB);
  unsigned short* kfr   = (unsigned short*)(ws + 80 * MB);
  unsigned short* kfi   = (unsigned short*)(ws + 88 * MB);
  unsigned short* vfr   = (unsigned short*)(ws + 96 * MB);
  float*          energ = (float*)(ws + 104 * MB);           // 16 KB
  float* att_f32 = qf32;
  float* ypre    = kf32;
  unsigned short* att_bf = xbf;

  const int NX = MM * CC;
  const int NW = CC * CC;

  energy_kernel<<<MM, 256, 0, stream>>>(x, energ);

  f32_to_bf16_kernel<<<(NX + 255) / 256, 256, 0, stream>>>(x,  xbf,  NX);
  f32_to_bf16_kernel<<<(NW + 255) / 256, 256, 0, stream>>>(wq, wbf0, NW);
  f32_to_bf16_kernel<<<(NW + 255) / 256, 256, 0, stream>>>(wk, wbf1, NW);
  f32_to_bf16_kernel<<<(NW + 255) / 256, 256, 0, stream>>>(wv, wbf2, NW);
  f32_to_bf16_kernel<<<(NW + 255) / 256, 256, 0, stream>>>(wo, wbf3, NW);

  dim3 ggrid(MM / 128, CC / 128);
  gemm_bf16_kernel<<<ggrid, 256, 0, stream>>>(xbf, wbf0, bq, qf32);
  gemm_bf16_kernel<<<ggrid, 256, 0, stream>>>(xbf, wbf1, bk, kf32);
  gemm_bf16_kernel<<<ggrid, 256, 0, stream>>>(xbf, wbf2, bv, vf32);

  fft_filter_kernel<<<BB * HH * HD, 128, 0, stream>>>(qf32, qfr, qfi, alpha, fas, fd);
  fft_filter_kernel<<<BB * HH * HD, 128, 0, stream>>>(kf32, kfr, kfi, alpha, fas, fd);
  fft_filter_kernel<<<BB * HH * HD, 128, 0, stream>>>(vf32, vfr, nullptr, alpha, fas, fd);

  flash_attn_kernel<<<dim3(BB * HH, TT / 128), 256, 0, stream>>>(
      qfr, qfi, kfr, kfi, vfr, att_f32);

  f32_to_bf16_kernel<<<(NX + 255) / 256, 256, 0, stream>>>(att_f32, att_bf, NX);
  gemm_bf16_kernel<<<ggrid, 256, 0, stream>>>(att_bf, wbf3, bo, ypre);

  final_scale_kernel<<<MM, 256, 0, stream>>>(ypre, energ, enrm, out);
}
